// GraphEncoder_3616362463821
// MI455X (gfx1250) — compile-verified, matrix operand order checked by reading, not run on hardware
//
#include <hip/hip_runtime.h>

// ---------------------------------------------------------------------------
// CDNA5 (gfx1250) GraphEncoder: embed -> 4x (WMMA GEMM -> edge scatter-add ->
// fused WMMA GRU) -> gather/concat -> WMMA output GEMM.
// GEMMs: v_wmma_f32_16x16x32_bf16, weights staged in LDS (ds_load_b128),
// 4 col-tiles per wave for A reuse. Scatter: f32 atomics into L2-resident agg
// with global_prefetch_b8 on the next gathered row.
// ---------------------------------------------------------------------------

typedef __attribute__((ext_vector_type(16))) __bf16 v16bf;
typedef __attribute__((ext_vector_type(8)))  __bf16 v8bf;
typedef __attribute__((ext_vector_type(8)))  float  v8f;

union BF16x16 { v16bf v; __bf16 e[16]; };
union BF16x8  { v8bf  v; __bf16 e[8];  };
union F32x8   { v8f   v; float  e[8];  };

#define WMMA_BF16(a, b, c) \
    __builtin_amdgcn_wmma_f32_16x16x32_bf16(false, (a), false, (b), (short)0, (c), false, false)

// A-fragment (16x32, MxK) for one lane from f32 row-major source.
// p0 = A + (row0 + (lane&15))*lda + 8*(lane>>4); lane holds K runs
// [kb,kb+8) and [16+kb,16+kb+8): two float4 pairs, cvt to bf16 in regs.
__device__ __forceinline__ v16bf load_frag_a_f32(const float* __restrict__ p0) {
    float4 x0 = *(const float4*)(p0 + 0);
    float4 x1 = *(const float4*)(p0 + 4);
    float4 x2 = *(const float4*)(p0 + 16);
    float4 x3 = *(const float4*)(p0 + 20);
    BF16x16 f;
    f.e[0]  = (__bf16)x0.x; f.e[1]  = (__bf16)x0.y; f.e[2]  = (__bf16)x0.z; f.e[3]  = (__bf16)x0.w;
    f.e[4]  = (__bf16)x1.x; f.e[5]  = (__bf16)x1.y; f.e[6]  = (__bf16)x1.z; f.e[7]  = (__bf16)x1.w;
    f.e[8]  = (__bf16)x2.x; f.e[9]  = (__bf16)x2.y; f.e[10] = (__bf16)x2.z; f.e[11] = (__bf16)x2.w;
    f.e[12] = (__bf16)x3.x; f.e[13] = (__bf16)x3.y; f.e[14] = (__bf16)x3.z; f.e[15] = (__bf16)x3.w;
    return f.v;
}

// B-fragment (32x16, KxN) for one lane from bf16 stored [outCol][K]
// (works for global or LDS pointers; inlined so addrspace is inferred and
// LDS sources lower to ds_load_b128).
__device__ __forceinline__ v16bf load_frag_b_bf16(const __bf16* p0) {
    BF16x8 lo, hi;
    lo.v = *(const v8bf*)(p0 + 0);
    hi.v = *(const v8bf*)(p0 + 16);
    BF16x16 f;
#pragma unroll
    for (int i = 0; i < 8; ++i) { f.e[i] = lo.e[i]; f.e[8 + i] = hi.e[i]; }
    return f.v;
}

// C/D layout store: lane = column, 8 VGPRs = rows (rows mrow..mrow+7).
__device__ __forceinline__ void store_tile(float* __restrict__ C, int ldc, int mrow,
                                           int col, v8f acc, float bv) {
    F32x8 r; r.v = acc;
#pragma unroll
    for (int v = 0; v < 8; ++v)
        C[(size_t)(mrow + v) * ldc + col] = r.e[v] + bv;
}

// ------------------------- prep kernels -----------------------------------

__global__ void k_cvt_bf16(const float* __restrict__ in, __bf16* __restrict__ out, int n) {
    int t = blockIdx.x * blockDim.x + threadIdx.x;
    if (t < n) out[t] = (__bf16)in[t];
}

// ggc_w[l][k][j] -> ggcT[l][j][k]  (4 x 256 x 256)
__global__ void k_tcvt_ggc(const float* __restrict__ in, __bf16* __restrict__ out) {
    int t = blockIdx.x * blockDim.x + threadIdx.x;
    if (t >= 4 * 256 * 256) return;
    int l = t >> 16, r = t & 65535, j = r >> 8, k = r & 255;
    out[t] = (__bf16)in[(l << 16) + (k << 8) + j];
}

__global__ void k_zero_f4(float4* __restrict__ p, int n4) {
    int t = blockIdx.x * blockDim.x + threadIdx.x;
    if (t < n4) p[t] = make_float4(0.f, 0.f, 0.f, 0.f);
}

// h[n][c] = emb[x[n]][c]
__global__ void k_embed(const int* __restrict__ x, const float* __restrict__ emb,
                        float* __restrict__ h, int n_nodes) {
    int t = blockIdx.x * blockDim.x + threadIdx.x;          // n_nodes * 64 float4s
    if (t >= n_nodes * 64) return;
    int n = t >> 6, c = (t & 63) << 2;
    *(float4*)(h + (size_t)n * 256 + c) =
        *(const float4*)(emb + (size_t)x[n] * 256 + c);
}

// ------------------------- LDS-staged WMMA GEMM ---------------------------
// C[M x Ncols] = A[M x K](f32) * Bt[Ncols x K](bf16) + bias.
// Block = 8 waves. blockIdx.y = 64-column group (B slice -> LDS once).
// Each wave: 4 row tiles x 4 col tiles; A frag loaded once per k-step and
// reused across the 4 WMMAs; B frags served from LDS.
__global__ void k_wmma_gemm(const float* __restrict__ A, const __bf16* __restrict__ Bt,
                            const float* __restrict__ bias, float* __restrict__ C,
                            int M, int Ncols, int K) {
    extern __shared__ __bf16 sB[];                       // [64][K]
    const int col0g = blockIdx.y * 64;

    // cooperative stage of the 64xK bf16 weight slice (contiguous in Bt)
    {
        const uint4* src = (const uint4*)(Bt + (size_t)col0g * K);
        uint4* dst = (uint4*)sB;
        const int n16 = (64 * K) >> 3;                   // 16B chunks
        for (int i = threadIdx.x; i < n16; i += blockDim.x) dst[i] = src[i];
    }
    __syncthreads();

    const int wave = threadIdx.x >> 5;
    const int lane = threadIdx.x & 31, lr = lane & 15, kb = (lane >> 4) << 3;
    const int mTiles = M >> 4;
    const int rowTile0 = blockIdx.x * 32 + wave;         // 8 waves x 4 row tiles

#pragma unroll
    for (int rt = 0; rt < 4; ++rt) {
        const int tm = rowTile0 + rt * 8;
        if (tm >= mTiles) continue;                      // wave-uniform
        const int row0 = tm << 4;
        v8f a0 = {}, a1 = {}, a2 = {}, a3 = {};
        const float*  ap = A + (size_t)(row0 + lr) * K + kb;
        const __bf16* bp = sB + (size_t)lr * K + kb;
        for (int k = 0; k < K; k += 32) {
            v16bf a = load_frag_a_f32(ap + k);
            v16bf b0 = load_frag_b_bf16(bp + k);
            v16bf b1 = load_frag_b_bf16(bp + 16 * K + k);
            v16bf b2 = load_frag_b_bf16(bp + 32 * K + k);
            v16bf b3 = load_frag_b_bf16(bp + 48 * K + k);
            a0 = WMMA_BF16(a, b0, a0);
            a1 = WMMA_BF16(a, b1, a1);
            a2 = WMMA_BF16(a, b2, a2);
            a3 = WMMA_BF16(a, b3, a3);
        }
        const int mrow = row0 + ((lane >> 4) << 3);
        const float bv0 = bias ? bias[col0g + 0  + lr] : 0.f;
        const float bv1 = bias ? bias[col0g + 16 + lr] : 0.f;
        const float bv2 = bias ? bias[col0g + 32 + lr] : 0.f;
        const float bv3 = bias ? bias[col0g + 48 + lr] : 0.f;
        store_tile(C, Ncols, mrow, col0g + 0  + lr, a0, bv0);
        store_tile(C, Ncols, mrow, col0g + 16 + lr, a1, bv1);
        store_tile(C, Ncols, mrow, col0g + 32 + lr, a2, bv2);
        store_tile(C, Ncols, mrow, col0g + 48 + lr, a3, bv3);
    }
}

// ------------------------- edge scatter-add -------------------------------
// 4 edges per wave; agg[dst] += m[src] * w (256 ch, 8/lane). agg and m are
// L2-resident (64MB each); prefetch the next source row while doing atomics.
__global__ void k_scatter(const float* __restrict__ m, const int* __restrict__ src,
                          const int* __restrict__ dst, const float* __restrict__ ew,
                          float* __restrict__ agg, int E) {
    const int e0 = ((blockIdx.x * blockDim.x + threadIdx.x) >> 5) << 2;
    if (e0 >= E) return;
    const int lane = threadIdx.x & 31;
    int s = src[e0];
#pragma unroll
    for (int i = 0; i < 4; ++i) {
        const int e = e0 + i;
        if (e >= E) break;                               // wave-uniform
        const int scur = s;
        if (i < 3 && e + 1 < E) {
            s = src[e + 1];
            __builtin_prefetch(m + (size_t)s * 256 + (lane << 3), 0, 0);
        }
        const int d = dst[e];
        const float w = ew[e];
        const float4* mp = (const float4*)(m + (size_t)scur * 256) + lane * 2;
        float4 a = mp[0], b = mp[1];
        float* ap = agg + (size_t)d * 256 + lane * 8;
        atomicAdd(ap + 0, a.x * w); atomicAdd(ap + 1, a.y * w);
        atomicAdd(ap + 2, a.z * w); atomicAdd(ap + 3, a.w * w);
        atomicAdd(ap + 4, b.x * w); atomicAdd(ap + 5, b.y * w);
        atomicAdd(ap + 6, b.z * w); atomicAdd(ap + 7, b.w * w);
    }
}

// ------------------------- fused WMMA GRU cell ----------------------------
// blockIdx.y = one 16-col tile; block stages all six gate-weight slabs
// (r/z/n x {w_ih,w_hh}, 6 x 16 x 256 bf16 = 48KB) in LDS, then 8 waves sweep
// 32 row tiles. Gate matrices (64000x768 x2) never touch memory.
__global__ void k_gru(const float* __restrict__ agg, const float* __restrict__ h,
                      const __bf16* __restrict__ wih, const __bf16* __restrict__ whh,
                      const float* __restrict__ bih, const float* __restrict__ bhh,
                      float* __restrict__ hout, int M) {
    extern __shared__ __bf16 sW[];                       // [6][16][256]
    const int col0 = blockIdx.y << 4;

    // stage 6 contiguous 16x256 slabs (8KB each)
    {
        uint4* dst = (uint4*)sW;
        const int slab16 = (16 * 256) >> 3;              // 512 16B chunks per slab
#pragma unroll
        for (int g = 0; g < 3; ++g) {
            const uint4* si = (const uint4*)(wih + (size_t)(g * 256 + col0) * 256);
            const uint4* sh = (const uint4*)(whh + (size_t)(g * 256 + col0) * 256);
            for (int i = threadIdx.x; i < slab16; i += blockDim.x) {
                dst[g * slab16 + i]       = si[i];
                dst[(3 + g) * slab16 + i] = sh[i];
            }
        }
    }
    __syncthreads();

    const int wave = threadIdx.x >> 5;
    const int lane = threadIdx.x & 31, lr = lane & 15, kb = (lane >> 4) << 3;
    const int mTiles = M >> 4;
    const int rowTile0 = blockIdx.x * 32 + wave;
    const __bf16* bp = sW + (size_t)lr * 256 + kb;
    const int jc = col0 + lr;
    const float b_ir = bih[jc], b_iz = bih[256 + jc], b_in = bih[512 + jc];
    const float b_hr = bhh[jc], b_hz = bhh[256 + jc], b_hn = bhh[512 + jc];

#pragma unroll
    for (int rt = 0; rt < 4; ++rt) {
        const int tm = rowTile0 + rt * 8;
        if (tm >= mTiles) continue;                      // wave-uniform
        const int row0 = tm << 4;
        v8f ir = {}, iz = {}, in_ = {}, hr = {}, hz = {}, hn = {};
        const float* agp = agg + (size_t)(row0 + lr) * 256 + kb;
        const float* hp  = h   + (size_t)(row0 + lr) * 256 + kb;
        for (int k = 0; k < 256; k += 32) {
            v16bf aa = load_frag_a_f32(agp + k);
            v16bf ah = load_frag_a_f32(hp + k);
            v16bf b;
            b = load_frag_b_bf16(bp + 0 * 4096 + k);  ir  = WMMA_BF16(aa, b, ir);
            b = load_frag_b_bf16(bp + 1 * 4096 + k);  iz  = WMMA_BF16(aa, b, iz);
            b = load_frag_b_bf16(bp + 2 * 4096 + k);  in_ = WMMA_BF16(aa, b, in_);
            b = load_frag_b_bf16(bp + 3 * 4096 + k);  hr  = WMMA_BF16(ah, b, hr);
            b = load_frag_b_bf16(bp + 4 * 4096 + k);  hz  = WMMA_BF16(ah, b, hz);
            b = load_frag_b_bf16(bp + 5 * 4096 + k);  hn  = WMMA_BF16(ah, b, hn);
        }
        const int mrow = row0 + ((lane >> 4) << 3);
        F32x8 fir, fiz, fin, fhr, fhz, fhn;
        fir.v = ir; fiz.v = iz; fin.v = in_; fhr.v = hr; fhz.v = hz; fhn.v = hn;
#pragma unroll
        for (int v = 0; v < 8; ++v) {
            const float hold = h[(size_t)(mrow + v) * 256 + jc];
            const float r  = 1.f / (1.f + __expf(-(fir.e[v] + b_ir + fhr.e[v] + b_hr)));
            const float z  = 1.f / (1.f + __expf(-(fiz.e[v] + b_iz + fhz.e[v] + b_hz)));
            const float nn = tanhf(fin.e[v] + b_in + r * (fhn.e[v] + b_hn));
            hout[(size_t)(mrow + v) * 256 + jc] = (1.f - z) * nn + z * hold;
        }
    }
}

// ------------------------- gather + concat --------------------------------
// cat[p][0:256] = h[gidx[p]] * mask[p]; cat[p][256:512] = encoder_embed[p]
__global__ void k_gather_concat(const float* __restrict__ h, const int* __restrict__ gidx,
                                const unsigned char* __restrict__ mask,
                                const float* __restrict__ enc, float* __restrict__ cat) {
    int t = blockIdx.x * blockDim.x + threadIdx.x;       // 8192 * 128 float4s
    if (t >= 8192 * 128) return;
    int row = t >> 7, c = (t & 127) << 2;
    float4 v;
    if (c < 256) {
        float mk = mask[row] ? 1.f : 0.f;
        float4 s = *(const float4*)(h + (size_t)gidx[row] * 256 + c);
        v = make_float4(s.x * mk, s.y * mk, s.z * mk, s.w * mk);
    } else {
        v = *(const float4*)(enc + (size_t)row * 256 + (c - 256));
    }
    *(float4*)(cat + (size_t)row * 512 + c) = v;
}

// ---------------------------------------------------------------------------

extern "C" void kernel_launch(void* const* d_in, const int* in_sizes, int n_in,
                              void* d_out, int out_size, void* d_ws, size_t ws_size,
                              hipStream_t stream) {
    (void)n_in; (void)out_size; (void)ws_size;
    const int*   x    = (const int*)d_in[0];
    const int*   ei   = (const int*)d_in[1];
    const float* ew   = (const float*)d_in[2];
    const int*   gidx = (const int*)d_in[3];
    const unsigned char* mask = (const unsigned char*)d_in[4];   // jnp bool = 1 byte
    const float* enc  = (const float*)d_in[5];
    const float* emb  = (const float*)d_in[6];
    const float* ggc  = (const float*)d_in[7];
    const float* wih  = (const float*)d_in[8];
    const float* whh  = (const float*)d_in[9];
    const float* bih  = (const float*)d_in[10];
    const float* bhh  = (const float*)d_in[11];
    const float* outw = (const float*)d_in[12];
    const float* outb = (const float*)d_in[13];
    float* out = (float*)d_out;

    const int Nn = in_sizes[0];              // 64000
    const int E  = in_sizes[1] / 2;          // 1,024,000
    const int H  = 256, BL = 16 * 512, CAT = 512, OUTC = 256;

    // Workspace carve (all 256B aligned). Total ~279 MB.
    char* ws = (char*)d_ws; size_t off = 0;
    auto carve = [&](size_t bytes) -> char* {
        char* p = ws + off; off = (off + bytes + 255) & ~(size_t)255; return p;
    };
    float*  hA   = (float*)carve((size_t)Nn * H * 4);
    float*  hB   = (float*)carve((size_t)Nn * H * 4);
    float*  m    = (float*)carve((size_t)Nn * H * 4);
    float*  agg  = (float*)carve((size_t)Nn * H * 4);
    float*  cat  = (float*)carve((size_t)BL * CAT * 4);
    __bf16* ggcT = (__bf16*)carve((size_t)4 * H * H * 2);
    __bf16* wihB = (__bf16*)carve((size_t)3 * H * H * 2);
    __bf16* whhB = (__bf16*)carve((size_t)3 * H * H * 2);
    __bf16* outwB = (__bf16*)carve((size_t)OUTC * CAT * 2);

    // --- weight prep (bf16, K-contiguous layouts; L2-resident thereafter) ---
    k_cvt_bf16<<<(3 * H * H + 255) / 256, 256, 0, stream>>>(wih, wihB, 3 * H * H);
    k_cvt_bf16<<<(3 * H * H + 255) / 256, 256, 0, stream>>>(whh, whhB, 3 * H * H);
    k_cvt_bf16<<<(OUTC * CAT + 255) / 256, 256, 0, stream>>>(outw, outwB, OUTC * CAT);
    k_tcvt_ggc<<<(4 * H * H + 255) / 256, 256, 0, stream>>>(ggc, ggcT);

    // --- embedding gather ---
    k_embed<<<(Nn * 64 + 255) / 256, 256, 0, stream>>>(x, emb, hA, Nn);

    float* hcur = hA; float* hnext = hB;
    const int mTiles    = Nn >> 4;                                 // 4000
    const int rowBlocks = (mTiles + 31) / 32;                      // 125
    const int zeroBlocks = (Nn * 64 + 255) / 256;
    const int scatBlocks = ((E + 3) / 4 * 32 + 255) / 256;
    const size_t smemGemmH = (size_t)64 * H * 2;                   // 32 KB
    const size_t smemGru   = (size_t)6 * 16 * 256 * 2;             // 48 KB
    for (int l = 0; l < 4; ++l) {
        k_wmma_gemm<<<dim3(rowBlocks, H / 64), 256, smemGemmH, stream>>>(
            hcur, ggcT + (size_t)l * H * H, nullptr, m, Nn, H, H);
        k_zero_f4<<<zeroBlocks, 256, 0, stream>>>((float4*)agg, Nn * 64);
        k_scatter<<<scatBlocks, 256, 0, stream>>>(m, ei, ei + E, ew, agg, E);
        k_gru<<<dim3(rowBlocks, H / 16), 256, smemGru, stream>>>(
            agg, hcur, wihB, whhB, bih, bhh, hnext, Nn);
        float* t = hcur; hcur = hnext; hnext = t;
    }

    // --- gather/concat + output GEMM ---
    k_gather_concat<<<(BL * 128 + 255) / 256, 256, 0, stream>>>(hcur, gidx, mask, enc, cat);
    const int outRowBlocks = ((BL >> 4) + 31) / 32;                // 16
    const size_t smemGemmO = (size_t)64 * CAT * 2;                 // 64 KB
    k_wmma_gemm<<<dim3(outRowBlocks, OUTC / 64), 256, smemGemmO, stream>>>(
        cat, outwB, outb, out, BL, OUTC, CAT);
}
